// MultiheadAttention_37495064494662
// MI455X (gfx1250) — matrix operand order repaired and checked by reading, not compiled, not run on hardware
//
#include <hip/hip_runtime.h>

typedef __attribute__((ext_vector_type(16))) __bf16 v16bf;
typedef __attribute__((ext_vector_type(8)))  float  v8f;
typedef __attribute__((ext_vector_type(4)))  float  v4f;
typedef __attribute__((ext_vector_type(4)))  unsigned int v4u;
typedef __attribute__((ext_vector_type(2)))  unsigned int v2u;

constexpr int Bb = 4, Ll = 1024, Ee = 2048, Hh = 16, Dd = 128;
constexpr int BL = Bb * Ll; // 4096 token rows
constexpr float SCL = 0.08838834764831845f; // 1/sqrt(128)

// Load a 16-element bf16 fragment as two 16B chunks at element offsets o0,o1.
__device__ inline v16bf ld_frag(const __bf16* base, int o0, int o1) {
  union { v16bf v; v4u q[2]; } u;
  u.q[0] = *(const v4u*)(base + o0);
  u.q[1] = *(const v4u*)(base + o1);
  return u.v;
}

__device__ inline v8f wmma_bf16(v16bf a, v16bf b, v8f c) {
  return __builtin_amdgcn_wmma_f32_16x16x32_bf16(false, a, false, b, (short)0, c,
                                                 false, false);
}

__device__ inline float red_max16(float v) {
  v = fmaxf(v, __shfl_xor(v, 1, 16));
  v = fmaxf(v, __shfl_xor(v, 2, 16));
  v = fmaxf(v, __shfl_xor(v, 4, 16));
  v = fmaxf(v, __shfl_xor(v, 8, 16));
  return v;
}
__device__ inline float red_sum16(float v) {
  v += __shfl_xor(v, 1, 16);
  v += __shfl_xor(v, 2, 16);
  v += __shfl_xor(v, 4, 16);
  v += __shfl_xor(v, 8, 16);
  return v;
}

// fp32 -> bf16 weight conversion (one-time streaming pass)
__global__ __launch_bounds__(256) void convw_kernel(
    const float* W0, const float* W1, const float* W2, const float* W3, __bf16* dst)
{
  const int z = blockIdx.y;
  const float* src = (z == 0) ? W0 : (z == 1) ? W1 : (z == 2) ? W2 : W3;
  __bf16* d = dst + (size_t)z * Ee * Ee;
  const size_t i = ((size_t)blockIdx.x * blockDim.x + threadIdx.x) * 8;
  v4f a = *(const v4f*)(src + i);
  v4f b = *(const v4f*)(src + i + 4);
  union { v4u u; __bf16 h[8]; } t;
  #pragma unroll
  for (int j = 0; j < 4; ++j) { t.h[j] = (__bf16)a[j]; t.h[4 + j] = (__bf16)b[j]; }
  *(v4u*)(d + i) = t.u;
}

struct LdsQKV {
  __bf16 As[2][128][40];
  __bf16 Bs[2][128][40];
  float rs[128];
  float X[32][132];
};
struct LdsOut {
  __bf16 As[2][128][40];
  __bf16 Bs[2][128][40];
};

// MODE 0: QKV projection (blockIdx.z selects q/k/v), fused bias+rmsnorm+rope epilogue.
// MODE 1: output projection from bf16 activations, fp32 output.
// B-tiles are staged with async LDS DMA (global_load_async_to_lds_b128), double-buffered.
template <int MODE>
__global__ __launch_bounds__(256) void gemm_fused(
    const float* A0, const float* A1, const float* A2, const __bf16* Abf,
    const __bf16* Wall,
    const float* b0, const float* b1, const float* b2,
    const float* cosp, const float* sinp,
    __bf16* qbf, __bf16* kbf, __bf16* vTbf, float* vf32, float* outf)
{
  __shared__ alignas(16) char smem_raw[(MODE == 0) ? sizeof(LdsQKV) : sizeof(LdsOut)];
  LdsQKV& S = *reinterpret_cast<LdsQKV*>(smem_raw);

  const int tid  = threadIdx.x;
  const int lane = tid & 31, wv = tid >> 5;
  const int wm = wv & 3, wn = wv >> 2;       // 4 waves along M, 2 along N
  const int lh = lane >> 4, ll = lane & 15;
  const int z  = (MODE == 0) ? (int)blockIdx.z : 0;
  const int n0 = blockIdx.x * 128;
  const int m0 = blockIdx.y * 128;
  const int head = blockIdx.x; // N tile == one head for MODE 0

  const float* Ap = (MODE == 0) ? ((z == 0) ? A0 : (z == 1) ? A1 : A2) : nullptr;
  const float* bp = (MODE == 0) ? ((z == 0) ? b0 : (z == 1) ? b1 : b2) : b0;
  const __bf16* Wp = Wall + (size_t)((MODE == 0) ? z : 3) * Ee * Ee + (size_t)n0 * Ee;
  const unsigned long long wbase = (unsigned long long)Wp;

  v8f acc[2][4];
  #pragma unroll
  for (int mt = 0; mt < 2; ++mt)
    #pragma unroll
    for (int nt = 0; nt < 4; ++nt)
      #pragma unroll
      for (int i = 0; i < 8; ++i) acc[mt][nt][i] = 0.0f;

  const int lr = tid >> 3;        // 0..31 row group for A staging
  const int k4 = (tid & 7) << 2;  // 0..28 within 32-wide k slab

  v4f aregf[4];
  v2u aregh[4];

  // Issue async DMA of one 128x32 bf16 B tile into Bs[buf]; 2 b128 chunks/thread.
  auto issue_b = [&](int buf, int kk) {
    #pragma unroll
    for (int i = 0; i < 2; ++i) {
      const int c   = tid + i * 256;     // 512 16B chunks
      const int row = c >> 2;
      const int k8  = (c & 3) * 8;
      const unsigned ldsoff =
          (unsigned)((const char*)&S.Bs[buf][row][k8] - (const char*)&S);
      const int goff = (row * Ee + kk + k8) * 2; // byte offset from wbase
      asm volatile("global_load_async_to_lds_b128 %0, %1, %2"
                   :: "v"(ldsoff), "v"(goff), "s"(wbase) : "memory");
    }
  };
  auto load_a = [&](int kk) {
    #pragma unroll
    for (int i = 0; i < 4; ++i) {
      const int row = lr + i * 32;
      if (MODE == 0)
        aregf[i] = *(const v4f*)(Ap + (size_t)(m0 + row) * Ee + kk + k4);
      else
        aregh[i] = *(const v2u*)(Abf + (size_t)(m0 + row) * Ee + kk + k4);
    }
  };
  auto store_a = [&](int buf) {
    #pragma unroll
    for (int i = 0; i < 4; ++i) {
      const int row = lr + i * 32;
      __bf16* d = &S.As[buf][row][k4];
      if (MODE == 0) {
        d[0] = (__bf16)aregf[i].x; d[1] = (__bf16)aregf[i].y;
        d[2] = (__bf16)aregf[i].z; d[3] = (__bf16)aregf[i].w;
      } else {
        *(v2u*)d = aregh[i];
      }
    }
  };

  // prologue: stage tile 0
  issue_b(0, 0);
  load_a(0);
  store_a(0);
  asm volatile("s_wait_asynccnt 0" ::: "memory");
  __syncthreads();

  int cb = 0;
  for (int kk = 0; kk < Ee; kk += 32) {
    const int nb = cb ^ 1;
    const bool more = (kk + 32 < Ee);
    if (more) {
      issue_b(nb, kk + 32);  // DMA next B tile, overlapped with WMMAs below
      load_a(kk + 32);       // next A tile into registers
    }

    v16bf af[2], bfv[4];
    #pragma unroll
    for (int mt = 0; mt < 2; ++mt)
      af[mt] = ld_frag(&S.As[cb][wm * 32 + mt * 16 + ll][0], lh * 8, lh * 8 + 16);
    #pragma unroll
    for (int nt = 0; nt < 4; ++nt)
      bfv[nt] = ld_frag(&S.Bs[cb][wn * 64 + nt * 16 + ll][0], lh * 8, lh * 8 + 16);
    #pragma unroll
    for (int mt = 0; mt < 2; ++mt)
      #pragma unroll
      for (int nt = 0; nt < 4; ++nt)
        acc[mt][nt] = wmma_bf16(af[mt], bfv[nt], acc[mt][nt]);

    if (more) store_a(nb);
    asm volatile("s_wait_asynccnt 0" ::: "memory");
    __syncthreads();
    cb = nb;
  }

  // bias add (bias depends only on column)
  float bias[4];
  #pragma unroll
  for (int nt = 0; nt < 4; ++nt) bias[nt] = bp[n0 + wn * 64 + nt * 16 + ll];
  #pragma unroll
  for (int mt = 0; mt < 2; ++mt)
    #pragma unroll
    for (int nt = 0; nt < 4; ++nt)
      #pragma unroll
      for (int r = 0; r < 8; ++r) acc[mt][nt][r] += bias[nt];

  if (MODE == 1) {
    #pragma unroll
    for (int mt = 0; mt < 2; ++mt)
      #pragma unroll
      for (int nt = 0; nt < 4; ++nt)
        #pragma unroll
        for (int r = 0; r < 8; ++r) {
          const int grow = m0 + wm * 32 + mt * 16 + r + 8 * lh;
          const int col  = n0 + wn * 64 + nt * 16 + ll;
          outf[(size_t)grow * Ee + col] = acc[mt][nt][r];
        }
    return;
  }

  if (z == 2) {
    // value: write fp32 (b,h,l,d) and bf16 transposed (b,h,d,l)
    #pragma unroll
    for (int mt = 0; mt < 2; ++mt)
      #pragma unroll
      for (int nt = 0; nt < 4; ++nt)
        #pragma unroll
        for (int r = 0; r < 8; ++r) {
          const int grow = m0 + wm * 32 + mt * 16 + r + 8 * lh;
          const int bidx = grow >> 10, pos = grow & (Ll - 1);
          const int dc = wn * 64 + nt * 16 + ll;
          const float val = acc[mt][nt][r];
          vf32[((size_t)(bidx * Hh + head) * Ll + pos) * Dd + dc] = val;
          vTbf[((size_t)(bidx * Hh + head) * Dd + dc) * Ll + pos] = (__bf16)val;
        }
    return;
  }

  // q/k: rmsnorm over head dim (full 128-col tile) then rotary
  if (tid < 128) S.rs[tid] = 0.0f;
  __syncthreads();
  #pragma unroll
  for (int mt = 0; mt < 2; ++mt)
    #pragma unroll
    for (int r = 0; r < 8; ++r) {
      float p = 0.0f;
      #pragma unroll
      for (int nt = 0; nt < 4; ++nt) { float v = acc[mt][nt][r]; p += v * v; }
      p = red_sum16(p);
      if (ll == 0) atomicAdd(&S.rs[wm * 32 + mt * 16 + r + 8 * lh], p);
    }
  __syncthreads();

  __bf16* dst = (z == 0) ? qbf : kbf;
  #pragma unroll
  for (int ph = 0; ph < 4; ++ph) {
    if (wm == ph) {
      #pragma unroll
      for (int mt = 0; mt < 2; ++mt)
        #pragma unroll
        for (int nt = 0; nt < 4; ++nt)
          #pragma unroll
          for (int r = 0; r < 8; ++r) {
            const int lrow = wm * 32 + mt * 16 + r + 8 * lh;
            const float scl = rsqrtf(S.rs[lrow] * (1.0f / 128.0f) + 1e-6f);
            S.X[lrow & 31][wn * 64 + nt * 16 + ll] = acc[mt][nt][r] * scl;
          }
    }
    __syncthreads();
    #pragma unroll
    for (int it = 0; it < 8; ++it) {
      const int idx = tid + it * 256;  // 32 rows x 64 pairs
      const int r = idx >> 6, c = idx & 63;
      const int grow = m0 + ph * 32 + r;
      const int bidx = grow >> 10, pos = grow & (Ll - 1);
      const float x1 = S.X[r][c], x2 = S.X[r][c + 64];
      const float co = cosp[pos * 64 + c], si = sinp[pos * 64 + c];
      const size_t base = ((size_t)(bidx * Hh + head) * Ll + pos) * Dd;
      dst[base + c]      = (__bf16)(x1 * co - x2 * si);
      dst[base + c + 64] = (__bf16)(x1 * si + x2 * co);
    }
    __syncthreads();
  }
}

// Flash-style causal attention + v-orthogonalization epilogue.
__global__ __launch_bounds__(128) void attn_kernel(
    const __bf16* __restrict__ qbf, const __bf16* __restrict__ kbf,
    const __bf16* __restrict__ vTbf, const float* __restrict__ vf32,
    __bf16* __restrict__ obf)
{
  __shared__ __bf16 pl[4][16][40];
  const int tid = threadIdx.x;
  const int lane = tid & 31, wv = tid >> 5;
  const int lh = lane >> 4, ll = lane & 15;
  const int bh = blockIdx.y;
  const int bb = bh >> 4, head = bh & 15;
  const int m0 = blockIdx.x * 64 + wv * 16;

  v16bf qf[4];
  const __bf16* qrow = qbf + ((size_t)bh * Ll + m0 + ll) * Dd;
  #pragma unroll
  for (int c = 0; c < 4; ++c)
    qf[c] = ld_frag(qrow, c * 32 + lh * 8, c * 32 + lh * 8 + 16);

  v8f o[8];
  #pragma unroll
  for (int c = 0; c < 8; ++c)
    #pragma unroll
    for (int i = 0; i < 8; ++i) o[c][i] = 0.0f;
  float mrow[8], lrow[8];
  #pragma unroll
  for (int r = 0; r < 8; ++r) { mrow[r] = -3.0e38f; lrow[r] = 0.0f; }

  const int nsteps = (m0 + 16 + 31) >> 5;
  for (int jt = 0; jt < nsteps; ++jt) {
    const int j = jt * 32;
    v8f sa[2];
    #pragma unroll
    for (int t = 0; t < 2; ++t) {
      #pragma unroll
      for (int i = 0; i < 8; ++i) sa[t][i] = 0.0f;
      const __bf16* krow = kbf + ((size_t)bh * Ll + j + t * 16 + ll) * Dd;
      #pragma unroll
      for (int c = 0; c < 4; ++c) {
        v16bf kf = ld_frag(krow, c * 32 + lh * 8, c * 32 + lh * 8 + 16);
        sa[t] = wmma_bf16(qf[c], kf, sa[t]);
      }
    }
    #pragma unroll
    for (int r = 0; r < 8; ++r) {
      const int row = m0 + r + 8 * lh;
      const int c0 = j + ll;
      const float v0 = (c0 <= row)      ? sa[0][r] * SCL : -3.0e38f;
      const float v1 = (c0 + 16 <= row) ? sa[1][r] * SCL : -3.0e38f;
      const float mx = red_max16(fmaxf(v0, v1));
      const float mnew = fmaxf(mrow[r], mx);
      const float alpha = __expf(mrow[r] - mnew);
      const float p0 = __expf(v0 - mnew);
      const float p1 = __expf(v1 - mnew);
      lrow[r] = lrow[r] * alpha + red_sum16(p0 + p1);
      mrow[r] = mnew;
      #pragma unroll
      for (int c = 0; c < 8; ++c) o[c][r] *= alpha;
      pl[wv][r + 8 * lh][ll]      = (__bf16)p0;  // C-layout -> LDS
      pl[wv][r + 8 * lh][16 + ll] = (__bf16)p1;
    }
    asm volatile("s_wait_dscnt 0" ::: "memory");   // order LDS transpose within wave
    v16bf pa = ld_frag(&pl[wv][ll][0], lh * 8, lh * 8 + 16);  // A-layout read
    #pragma unroll
    for (int c = 0; c < 8; ++c) {
      v16bf vf = ld_frag(vTbf + ((size_t)bh * Dd + c * 16 + ll) * Ll + j,
                         lh * 8, lh * 8 + 16);
      o[c] = wmma_bf16(pa, vf, o[c]);
    }
  }

  #pragma unroll
  for (int r = 0; r < 8; ++r) {
    const float li = 1.0f / lrow[r];
    #pragma unroll
    for (int c = 0; c < 8; ++c) o[c][r] *= li;
  }

  // o = o - (o . v_hat) v_hat  with v_hat = v / max(||v||, eps)
  float sov[8], svv[8];
  #pragma unroll
  for (int r = 0; r < 8; ++r) { sov[r] = 0.0f; svv[r] = 0.0f; }
  #pragma unroll
  for (int c = 0; c < 8; ++c)
    #pragma unroll
    for (int r = 0; r < 8; ++r) {
      const int row = m0 + r + 8 * lh;
      const float vv = vf32[((size_t)bh * Ll + row) * Dd + c * 16 + ll];
      sov[r] += o[c][r] * vv;
      svv[r] += vv * vv;
    }
  float fr[8];
  #pragma unroll
  for (int r = 0; r < 8; ++r) {
    const float so = red_sum16(sov[r]);
    const float sv = red_sum16(svv[r]);
    fr[r] = so / fmaxf(sv, 1e-18f);
  }
  #pragma unroll
  for (int c = 0; c < 8; ++c)
    #pragma unroll
    for (int r = 0; r < 8; ++r) {
      const int row = m0 + r + 8 * lh;
      const float vv = vf32[((size_t)bh * Ll + row) * Dd + c * 16 + ll];
      const float val = o[c][r] - fr[r] * vv;
      obf[((size_t)bb * Ll + row) * Ee + head * Dd + c * 16 + ll] = (__bf16)val;
    }
}

extern "C" void kernel_launch(void* const* d_in, const int* in_sizes, int n_in,
                              void* d_out, int out_size, void* d_ws, size_t ws_size,
                              hipStream_t stream) {
  const float* query = (const float*)d_in[0];
  const float* keyt  = (const float*)d_in[1];
  const float* value = (const float*)d_in[2];
  // d_in[3] attn_mask: causal, applied analytically
  const float* cosp  = (const float*)d_in[4];
  const float* sinp  = (const float*)d_in[5];
  const float* Wq = (const float*)d_in[6];  const float* bq = (const float*)d_in[7];
  const float* Wk = (const float*)d_in[8];  const float* bk = (const float*)d_in[9];
  const float* Wv = (const float*)d_in[10]; const float* bv = (const float*)d_in[11];
  const float* Wo = (const float*)d_in[12]; const float* bo = (const float*)d_in[13];
  float* out = (float*)d_out;

  char* ws = (char*)d_ws;
  const size_t qe = (size_t)Bb * Hh * Ll * Dd;  // 8,388,608 elements
  __bf16* qbf  = (__bf16*)(ws);                 // 16 MB
  __bf16* kbf  = (__bf16*)(ws + qe * 2);        // 16 MB
  __bf16* vTbf = (__bf16*)(ws + qe * 4);        // 16 MB
  float*  vf32 = (float*) (ws + qe * 6);        // 32 MB
  __bf16* obf  = (__bf16*)(ws + qe * 10);       // 16 MB
  __bf16* wbf  = (__bf16*)(ws + qe * 12);       // 32 MB (Wq,Wk,Wv,Wo bf16)

  convw_kernel<<<dim3((Ee * Ee) / (256 * 8), 4), 256, 0, stream>>>(Wq, Wk, Wv, Wo, wbf);
  gemm_fused<0><<<dim3(Ee / 128, BL / 128, 3), 256, 0, stream>>>(
      query, keyt, value, nullptr, wbf, bq, bk, bv, cosp, sinp,
      qbf, kbf, vTbf, vf32, nullptr);
  attn_kernel<<<dim3(Ll / 64, Bb * Hh), 128, 0, stream>>>(qbf, kbf, vTbf, vf32, obf);
  gemm_fused<1><<<dim3(Ee / 128, BL / 128, 1), 256, 0, stream>>>(
      nullptr, nullptr, nullptr, obf, wbf, bo, nullptr, nullptr,
      nullptr, nullptr, nullptr, nullptr, nullptr, nullptr, out);
}
